// MultiHeadAttention_63806034149862
// MI455X (gfx1250) — matrix operand order repaired and checked by reading, not compiled, not run on hardware
//
#include <hip/hip_runtime.h>

namespace {
constexpr int NH  = 16;
constexpr int HD  = 64;
constexpr int EMB = 1024;
constexpr int SEQ = 4096;
constexpr int NB  = 2;
}

typedef __attribute__((ext_vector_type(16))) __bf16 v16bf;
typedef __attribute__((ext_vector_type(8)))  __bf16 v8bf;
typedef __attribute__((ext_vector_type(8)))  float  v8f;
typedef __attribute__((ext_vector_type(4)))  float  v4f;
typedef __attribute__((ext_vector_type(4)))  unsigned int v4u;
typedef __attribute__((ext_vector_type(8)))  int v8i;
typedef __attribute__((ext_vector_type(4)))  int v4i;

union V16U { v16bf v; v8bf h[2]; };

__device__ __forceinline__ v8f wmma_bf16(v16bf a, v16bf b, v8f c) {
  // v_wmma_f32_16x16x32_bf16: D = A(16x32) x B(32x16) + C
  return __builtin_amdgcn_wmma_f32_16x16x32_bf16(false, a, false, b, (short)0, c, false, false);
}

// A-fragment (16x32, lane = M-row): lane<16 -> k = colbase+{0..7,16..23},
// lane>=16 -> k = colbase+{8..15,24..31}. Two 16B loads.
__device__ __forceinline__ v16bf frag_a(const __bf16* p0, int stride, int row, int colbase, int lane) {
  const __bf16* p = p0 + (long)row * stride + colbase + ((lane >> 4) << 3);
  V16U u;
  u.h[0] = *(const v8bf*)p;
  u.h[1] = *(const v8bf*)(p + 16);
  return u.v;
}

// B-fragment (32x16, lane = N-row): lane<16 -> k = colbase+{0..7,16..23},
// lane>=16 -> k = colbase+{8..15,24..31}. Two 16B loads.
__device__ __forceinline__ v16bf frag_b(const __bf16* p0, int stride, int row, int colbase, int lane) {
  const __bf16* p = p0 + (long)row * stride + colbase + ((lane >> 4) << 3);
  V16U u;
  u.h[0] = *(const v8bf*)p;
  u.h[1] = *(const v8bf*)(p + 16);
  return u.v;
}

// ---------------------------------------------------------------------------
// Tensor Data Mover: DMA a 2-D bf16 tile (tile_h rows x tile_w elems, global row
// stride in elems) into LDS, adding pad_amount DWORDs of LDS padding every
// 2^pad_interval*2 DWORDs (ISA cdna5 §8: D# groups; data_size=1 -> 2 bytes).
// Descriptor is uniform; issue from one wave, tracked by TENSORcnt.
// ---------------------------------------------------------------------------
__device__ __forceinline__ void tdm_load_tile_bf16(
    unsigned int lds_addr, unsigned long gaddr,
    unsigned int tensor_w, unsigned int tensor_h,
    unsigned int tile_w, unsigned int tile_h,
    unsigned long row_stride,
    unsigned int pad_interval_code, unsigned int pad_amount_code)
{
  v4u g0;
  g0[0] = 1u;                                            // count=1, user descriptor
  g0[1] = lds_addr;                                      // lds_addr [63:32]
  g0[2] = (unsigned int)(gaddr & 0xFFFFFFFFu);           // global_addr [95:64]
  g0[3] = (unsigned int)((gaddr >> 32) & 0x01FFFFFFu)    // global_addr [120:96]
        | (2u << 30);                                    // type=2 ("image")
  v8i g1;
  g1[0] = (int)((1u << 16)                               // data_size: 1 -> 2 bytes
        | (1u << 20)                                     // pad_enable
        | (pad_interval_code << 22)
        | (pad_amount_code << 25));
  g1[1] = (int)((tensor_w & 0xFFFFu) << 16);             // tensor_dim0[15:0] @ bits 63:48
  g1[2] = (int)(((tensor_w >> 16) & 0xFFFFu)             // tensor_dim0[31:16]
        | ((tensor_h & 0xFFFFu) << 16));                 // tensor_dim1[15:0]
  g1[3] = (int)(((tensor_h >> 16) & 0xFFFFu)             // tensor_dim1[31:16]
        | ((tile_w & 0xFFFFu) << 16));                   // tile_dim0
  g1[4] = (int)(tile_h & 0xFFFFu);                       // tile_dim1 (tile_dim2 = 0)
  g1[5] = (int)(unsigned int)(row_stride & 0xFFFFFFFFu); // tensor_dim0_stride[31:0]
  g1[6] = (int)((row_stride >> 32) & 0xFFFFu);           // stride[47:32], dim1_stride lo = 0
  g1[7] = 0;
  v4i z4 = {0, 0, 0, 0};
#if defined(__clang_major__) && (__clang_major__ >= 23)
  v8i z8 = {0, 0, 0, 0, 0, 0, 0, 0};
  __builtin_amdgcn_tensor_load_to_lds(g0, g1, z4, z4, z8, 0);
#else
  __builtin_amdgcn_tensor_load_to_lds(g0, g1, z4, z4, 0);
#endif
}

// Cooperative 128x32 fp32 tile -> bf16 LDS tile (row stride 40 elems = 80B)
__device__ __forceinline__ void coop_f32_tile(__bf16* lds, const float* src, long row0,
                                              int src_stride, int col0, int tid) {
  const int r = tid >> 1;
  const int c = (tid & 1) << 4;
  const float* g = src + (row0 + r) * (long)src_stride + col0 + c;
  v4f f0 = *(const v4f*)(g);
  v4f f1 = *(const v4f*)(g + 4);
  v4f f2 = *(const v4f*)(g + 8);
  v4f f3 = *(const v4f*)(g + 12);
  v8bf b0, b1;
  for (int i = 0; i < 4; ++i) {
    b0[i]     = (__bf16)f0[i];
    b0[i + 4] = (__bf16)f1[i];
    b1[i]     = (__bf16)f2[i];
    b1[i + 4] = (__bf16)f3[i];
  }
  __bf16* d = lds + r * 40 + c;
  *(v8bf*)d = b0;
  *(v8bf*)(d + 8) = b1;
}

__device__ __forceinline__ void coop_bf16_tile(__bf16* lds, const __bf16* src, long row0,
                                               int src_stride, int col0, int tid) {
  const int r = tid >> 1;
  const int c = (tid & 1) << 4;
  const __bf16* g = src + (row0 + r) * (long)src_stride + col0 + c;
  v8bf b0 = *(const v8bf*)g;
  v8bf b1 = *(const v8bf*)(g + 8);
  __bf16* d = lds + r * 40 + c;
  *(v8bf*)d = b0;
  *(v8bf*)(d + 8) = b1;
}

// ---------------------------------------------------------------------------
// Kernel 1: fused QKV projections.  D(feature, token) = W x X^T (+ bias).
// z = 0: Q (pre-scaled by 1/sqrt(64)*log2e, stored [b][h][s][d] bf16)
// z = 1: K (stored [b][h][s][d] bf16)
// z = 2: V (stored transposed [b][h][d][s] bf16 so PV B-fragments are contiguous)
// ---------------------------------------------------------------------------
__global__ __launch_bounds__(256) void mha_qkv_proj(
    const float* __restrict__ xq, const float* __restrict__ xk, const float* __restrict__ xv,
    const float* __restrict__ Wq, const float* __restrict__ bq,
    const float* __restrict__ Wk, const float* __restrict__ bk,
    const float* __restrict__ Wv, const float* __restrict__ bv,
    __bf16* __restrict__ Qg, __bf16* __restrict__ Kg, __bf16* __restrict__ Vt)
{
  __shared__ __bf16 Wl[128 * 40];
  __shared__ __bf16 Xl[128 * 40];
  const int z = blockIdx.z;
  const float* X    = (z == 0) ? xq : (z == 1) ? xk : xv;
  const float* W    = (z == 0) ? Wq : (z == 1) ? Wk : Wv;
  const float* bias = (z == 0) ? bq : (z == 1) ? bk : bv;
  const int tid = threadIdx.x, lane = tid & 31, wid = tid >> 5;
  const int wm = wid & 1, wn = wid >> 1;          // 2 x 4 wave grid over 128x128 tile
  const long tbase = (long)blockIdx.x * 128;      // token tile
  const int fbase = blockIdx.y * 128;             // feature tile

  v8f acc[4][2];
  for (int mt = 0; mt < 4; ++mt)
    for (int nt = 0; nt < 2; ++nt)
      for (int r = 0; r < 8; ++r) acc[mt][nt][r] = 0.f;

  for (int kk = 0; kk < EMB; kk += 32) {
    coop_f32_tile(Wl, W, fbase, EMB, kk, tid);
    coop_f32_tile(Xl, X, tbase, EMB, kk, tid);
    __syncthreads();
    v16bf af[4], bfr[2];
    for (int mt = 0; mt < 4; ++mt)
      af[mt] = frag_a(Wl, 40, wm * 64 + mt * 16 + (lane & 15), 0, lane);
    for (int nt = 0; nt < 2; ++nt)
      bfr[nt] = frag_b(Xl, 40, wn * 32 + nt * 16 + (lane & 15), 0, lane);
    for (int mt = 0; mt < 4; ++mt)
      for (int nt = 0; nt < 2; ++nt)
        acc[mt][nt] = wmma_bf16(af[mt], bfr[nt], acc[mt][nt]);
    __syncthreads();
  }

  const float QSCALE = 0.125f * 1.44269504088896340736f;  // (1/sqrt(HD)) * log2(e)
  for (int mt = 0; mt < 4; ++mt) {
    const int f0 = fbase + wm * 64 + mt * 16 + ((lane >> 4) << 3);  // 8 consecutive features
    const v4f bi0 = *(const v4f*)(bias + f0);
    const v4f bi1 = *(const v4f*)(bias + f0 + 4);
    const int hh = f0 >> 6;   // head
    const int d0 = f0 & 63;   // dim within head
    for (int nt = 0; nt < 2; ++nt) {
      const long t = tbase + wn * 32 + nt * 16 + (lane & 15);
      const long bb = t >> 12, ss = t & (SEQ - 1);
      float vals[8];
      for (int r = 0; r < 8; ++r)
        vals[r] = acc[mt][nt][r] + ((r < 4) ? bi0[r] : bi1[r - 4]);
      if (z == 0)
        for (int r = 0; r < 8; ++r) vals[r] *= QSCALE;
      if (z != 2) {
        v8bf o;
        for (int r = 0; r < 8; ++r) o[r] = (__bf16)vals[r];
        __bf16* dst = ((z == 0) ? Qg : Kg) + (((bb * NH + hh) * SEQ + ss) * HD + d0);
        *(v8bf*)dst = o;
      } else {
        for (int r = 0; r < 8; ++r)
          Vt[((bb * NH + hh) * HD + d0 + r) * (long)SEQ + ss] = (__bf16)vals[r];
      }
    }
  }
}

// ---------------------------------------------------------------------------
// Kernel 2: flash attention.  Scores computed transposed (S^T = K x Q^T) so the
// exp'd probabilities pack in-register into the PV A-fragment (no transpose).
// K/V tiles DMA'd to LDS by the Tensor Data Mover (wave 0), double-buffered so
// the next tile's DMA overlaps the current tile's WMMAs; TENSORcnt-tracked.
// ---------------------------------------------------------------------------
__global__ __launch_bounds__(128) void mha_flash_attn(
    const __bf16* __restrict__ Qg, const __bf16* __restrict__ Kg,
    const __bf16* __restrict__ Vt, __bf16* __restrict__ ctx)
{
  __shared__ __bf16 Kl[2][32 * 72];   // 32 keys x 64 d (TDM-padded rows: 144B)
  __shared__ __bf16 Vl[2][64 * 40];   // 64 d x 32 keys (TDM-padded rows: 80B)
  const int tid = threadIdx.x, lane = tid & 31, wv = tid >> 5;
  const int head = blockIdx.y, batch = blockIdx.z;
  const long bh = (long)batch * NH + head;
  const long qbase = (long)blockIdx.x * 64 + wv * 16;
  const __bf16* Qh = Qg + bh * SEQ * HD;
  const __bf16* Kh = Kg + bh * SEQ * HD;
  const __bf16* Vh = Vt + bh * HD * SEQ;
  const unsigned int kl_lds = (unsigned int)(unsigned long)(const void*)&Kl[0][0];
  const unsigned int vl_lds = (unsigned int)(unsigned long)(const void*)&Vl[0][0];

  v16bf qf[2];                                   // Q as B-operand, two 32-d halves
  for (int hh = 0; hh < 2; ++hh)
    qf[hh] = frag_b(Qh, HD, (int)(qbase + (lane & 15)), hh * 32, lane);

  float mrow = -1e30f, lrow = 0.f;               // per-query running max / sum
  v8f o[4];
  for (int dt = 0; dt < 4; ++dt)
    for (int r = 0; r < 8; ++r) o[dt][r] = 0.f;
  v8f zf;
  for (int r = 0; r < 8; ++r) zf[r] = 0.f;
  const int sb = (lane >> 4) << 3;

  if (wv == 0) {  // prologue DMA: tile 0 into buffer 0
    tdm_load_tile_bf16(kl_lds, (unsigned long)(const void*)Kh,
                       HD, SEQ, HD, 32, HD, 4, 3);        // 32x64, pad 128B->+16B
    tdm_load_tile_bf16(vl_lds, (unsigned long)(const void*)Vh,
                       SEQ, HD, 32, HD, SEQ, 3, 3);       // 64x32, pad 64B->+16B
  }

  for (int kb = 0; kb < SEQ; kb += 32) {
    const int buf = (kb >> 5) & 1;
    if (wv == 0) {
      if (kb + 32 < SEQ) {  // DMA next tile into the other buffer, then wait for current
        const int nb = buf ^ 1;
        tdm_load_tile_bf16(kl_lds + (unsigned int)(nb * 32 * 72 * 2),
                           (unsigned long)(const void*)(Kh + (long)(kb + 32) * HD),
                           HD, SEQ, HD, 32, HD, 4, 3);
        tdm_load_tile_bf16(vl_lds + (unsigned int)(nb * 64 * 40 * 2),
                           (unsigned long)(const void*)(Vh + (kb + 32)),
                           SEQ, HD, 32, HD, SEQ, 3, 3);
        __builtin_amdgcn_s_wait_tensorcnt(2);   // current tile done, next in flight
      } else {
        __builtin_amdgcn_s_wait_tensorcnt(0);
      }
    }
    __syncthreads();

    v16bf kf[2][2], vf[4];
    for (int t = 0; t < 2; ++t)
      for (int hh = 0; hh < 2; ++hh)
        kf[t][hh] = frag_a(&Kl[buf][0], 72, t * 16 + (lane & 15), hh * 32, lane);
    for (int dt = 0; dt < 4; ++dt)
      vf[dt] = frag_b(&Vl[buf][0], 40, dt * 16 + (lane & 15), 0, lane);

    // S^T tiles: C layout lane = query, regs = key (Q already scaled by log2e/8)
    v8f s0 = wmma_bf16(kf[0][0], qf[0], zf);
    s0 = wmma_bf16(kf[0][1], qf[1], s0);
    v8f s1 = wmma_bf16(kf[1][0], qf[0], zf);
    s1 = wmma_bf16(kf[1][1], qf[1], s1);

    float bm = -1e30f;
    for (int r = 0; r < 8; ++r) {
      bm = fmaxf(bm, s0[r]);
      bm = fmaxf(bm, s1[r]);
    }
    bm = fmaxf(bm, __shfl_xor(bm, 16, 32));      // combine the two key half-lanes
    const float mnew = fmaxf(mrow, bm);
    const float corr = __builtin_amdgcn_exp2f(mrow - mnew);

    v16bf pf;                                    // packs directly into PV A-fragment
    float rs = 0.f;
    for (int r = 0; r < 8; ++r) {
      const float p0 = __builtin_amdgcn_exp2f(s0[r] - mnew);
      const float p1 = __builtin_amdgcn_exp2f(s1[r] - mnew);
      rs += p0 + p1;
      pf[r]     = (__bf16)p0;
      pf[r + 8] = (__bf16)p1;
    }
    rs += __shfl_xor(rs, 16, 32);
    lrow = lrow * corr + rs;
    mrow = mnew;

    float cr[8];                                 // redistribute corr to O layout (regs = query)
    for (int r = 0; r < 8; ++r) cr[r] = __shfl(corr, sb + r, 32);
    for (int dt = 0; dt < 4; ++dt)
      for (int r = 0; r < 8; ++r) o[dt][r] *= cr[r];
    for (int dt = 0; dt < 4; ++dt)
      o[dt] = wmma_bf16(pf, vf[dt], o[dt]);
    __syncthreads();   // all reads of buf done before TDM may overwrite it
  }

  float li[8];
  for (int r = 0; r < 8; ++r) li[r] = 1.f / __shfl(lrow, sb + r, 32);
  for (int dt = 0; dt < 4; ++dt) {
    const int d = dt * 16 + (lane & 15);
    for (int r = 0; r < 8; ++r) {
      const long q = qbase + sb + r;
      ctx[((long)batch * SEQ + q) * EMB + head * HD + d] = (__bf16)(o[dt][r] * li[r]);
    }
  }
}

// ---------------------------------------------------------------------------
// Kernel 3: output projection, fp32 result: out = ctx @ Wo^T + bo
// ---------------------------------------------------------------------------
__global__ __launch_bounds__(256) void mha_out_proj(
    const __bf16* __restrict__ ctx, const float* __restrict__ Wo,
    const float* __restrict__ bo, float* __restrict__ out)
{
  __shared__ __bf16 Wl[128 * 40];
  __shared__ __bf16 Xl[128 * 40];
  const int tid = threadIdx.x, lane = tid & 31, wid = tid >> 5;
  const int wm = wid & 1, wn = wid >> 1;
  const long tbase = (long)blockIdx.x * 128;
  const int fbase = blockIdx.y * 128;

  v8f acc[4][2];
  for (int mt = 0; mt < 4; ++mt)
    for (int nt = 0; nt < 2; ++nt)
      for (int r = 0; r < 8; ++r) acc[mt][nt][r] = 0.f;

  for (int kk = 0; kk < EMB; kk += 32) {
    coop_f32_tile(Wl, Wo, fbase, EMB, kk, tid);
    coop_bf16_tile(Xl, ctx, tbase, EMB, kk, tid);
    __syncthreads();
    v16bf af[4], bfr[2];
    for (int mt = 0; mt < 4; ++mt)
      af[mt] = frag_a(Wl, 40, wm * 64 + mt * 16 + (lane & 15), 0, lane);
    for (int nt = 0; nt < 2; ++nt)
      bfr[nt] = frag_b(Xl, 40, wn * 32 + nt * 16 + (lane & 15), 0, lane);
    for (int mt = 0; mt < 4; ++mt)
      for (int nt = 0; nt < 2; ++nt)
        acc[mt][nt] = wmma_bf16(af[mt], bfr[nt], acc[mt][nt]);
    __syncthreads();
  }

  for (int mt = 0; mt < 4; ++mt) {
    const int f0 = fbase + wm * 64 + mt * 16 + ((lane >> 4) << 3);
    const v4f bi0 = *(const v4f*)(bo + f0);
    const v4f bi1 = *(const v4f*)(bo + f0 + 4);
    for (int nt = 0; nt < 2; ++nt) {
      const long t = tbase + wn * 32 + nt * 16 + (lane & 15);
      v4f o0, o1;
      for (int r = 0; r < 4; ++r) {
        o0[r] = acc[mt][nt][r] + bi0[r];
        o1[r] = acc[mt][nt][r + 4] + bi1[r];
      }
      float* dst = out + t * EMB + f0;
      *(v4f*)dst = o0;
      *(v4f*)(dst + 4) = o1;
    }
  }
}

extern "C" void kernel_launch(void* const* d_in, const int* in_sizes, int n_in,
                              void* d_out, int out_size, void* d_ws, size_t ws_size,
                              hipStream_t stream)
{
  (void)in_sizes; (void)n_in; (void)out_size; (void)ws_size;
  const float* xq = (const float*)d_in[0];
  const float* xk = (const float*)d_in[1];
  const float* xv = (const float*)d_in[2];
  const float* Wq = (const float*)d_in[3];
  const float* bq = (const float*)d_in[4];
  const float* Wk = (const float*)d_in[5];
  const float* bk = (const float*)d_in[6];
  const float* Wv = (const float*)d_in[7];
  const float* bv = (const float*)d_in[8];
  const float* Wo = (const float*)d_in[9];
  const float* bo = (const float*)d_in[10];

  const size_t T = (size_t)NB * SEQ * EMB;       // 8.4M elems per tensor
  __bf16* Qg  = (__bf16*)d_ws;                   // [b][h][s][d], pre-scaled
  __bf16* Kg  = Qg + T;                          // [b][h][s][d]
  __bf16* Vt  = Kg + T;                          // [b][h][d][s] (transposed)
  __bf16* ctx = Vt + T;                          // [b][s][e]

  mha_qkv_proj<<<dim3(SEQ * NB / 128, EMB / 128, 3), 256, 0, stream>>>(
      xq, xk, xv, Wq, bq, Wk, bk, Wv, bv, Qg, Kg, Vt);
  mha_flash_attn<<<dim3(SEQ / 64, NH, NB), 128, 0, stream>>>(Qg, Kg, Vt, ctx);
  mha_out_proj<<<dim3(SEQ * NB / 128, EMB / 128, 1), 256, 0, stream>>>(
      ctx, Wo, bo, (float*)d_out);
}
